// deformable_cross_grd_attention_42769284333656
// MI455X (gfx1250) — matrix operand order repaired and checked by reading, not compiled, not run on hardware
//
#include <hip/hip_runtime.h>
#include <hip/hip_bf16.h>

// ---------------------------------------------------------------------------
// Deformable cross-grid attention (MSDeformAttn + FFN) for MI455X / gfx1250.
// fp32 storage, bf16 V_WMMA_F32_16X16X32_BF16 compute, fused epilogues.
// ---------------------------------------------------------------------------

#define EMBED    256
#define HEADS    8
#define HEAD_DIM 32
#define LEVELS   4
#define POINTS   4
#define NQ       16384
#define HIDDEN   512
#define NVTOT    5440   // 64*64 + 32*32 + 16*16 + 8*8

typedef __attribute__((ext_vector_type(16))) __bf16 v16bf;
typedef __attribute__((ext_vector_type(8)))  float  v8f;

union Frag16 {
    v16bf v;
    uint4 q[2];
};

__device__ __forceinline__ unsigned short f2bf(float f) {
    unsigned int u = __float_as_uint(f);
    unsigned int r = u + 0x7FFFu + ((u >> 16) & 1u);   // round-to-nearest-even
    return (unsigned short)(r >> 16);
}

// ===========================================================================
// FAST PATH: fully tile-aligned GEMM (M%128==0, N%128==0, K%32==0).
//   C = act( A (MxK, lda=K) * W (KxN) + bias + rscale*R )
// Block tile 128x128x32, 8 waves; wave tile 32x64 = 2x4 WMMA frags.
// Vectorized staging (b128 global, b64/b128 LDS), software-pipelined K loop.
// ===========================================================================
__global__ __launch_bounds__(256) void wmma_gemm_aligned_kernel(
    const float* __restrict__ A, int lda,
    const float* __restrict__ W,
    const float* __restrict__ bias,
    const float* __restrict__ R, float rscale,
    float* __restrict__ C,
    int M, int N, int K, int relu)
{
    constexpr int BM = 128, BN = 128, BK = 32;
    __shared__ unsigned short As[BM * BK];   // [m][k]
    __shared__ unsigned short Bs[BN * BK];   // [n][k] (transposed in registers)

    const int tid  = threadIdx.x;
    const int m0   = blockIdx.x * BM;
    const int n0   = blockIdx.y * BN;
    const int wave = tid >> 5;
    const int lane = tid & 31;
    const int wm   = wave & 3;           // 4 groups of 32 rows
    const int wn   = wave >> 2;          // 2 groups of 64 cols
    const int l15  = lane & 15;
    const int lhalf = (lane >> 4) & 1;

    // A staging map: thread -> (row tid>>3 + 32*i, float4 col tid&7)
    const int arow = tid >> 3;           // 0..31
    const int af4  = (tid & 7) * 4;      // k offset 0,4,..,28
    // B staging map: thread -> column n = tid&127, k-half (tid>>7)*16
    const int bn   = tid & 127;
    const int bkh  = (tid >> 7) * 16;

    v8f acc[2][4];
#pragma unroll
    for (int i = 0; i < 2; ++i)
#pragma unroll
        for (int j = 0; j < 4; ++j)
            acc[i][j] = (v8f){0.f, 0.f, 0.f, 0.f, 0.f, 0.f, 0.f, 0.f};

    float4 aR[4];
    float  bR[16];

    // ---- prologue global load (k0 = 0) ----
#pragma unroll
    for (int i = 0; i < 4; ++i)
        aR[i] = *reinterpret_cast<const float4*>(
            &A[(size_t)(m0 + arow + 32 * i) * lda + af4]);
    {
        const float* wc = &W[(size_t)bkh * N + n0 + bn];
#pragma unroll
        for (int j = 0; j < 16; ++j) bR[j] = wc[(size_t)j * N];
    }

    for (int k0 = 0; k0 < K; k0 += BK) {
        // ---- stage regs -> LDS (bf16) ----
#pragma unroll
        for (int i = 0; i < 4; ++i) {
            ushort4 h;
            h.x = f2bf(aR[i].x); h.y = f2bf(aR[i].y);
            h.z = f2bf(aR[i].z); h.w = f2bf(aR[i].w);
            *reinterpret_cast<ushort4*>(&As[(arow + 32 * i) * BK + af4]) = h;
        }
        {
            unsigned short hb[16];
#pragma unroll
            for (int j = 0; j < 16; ++j) hb[j] = f2bf(bR[j]);
            *reinterpret_cast<uint4*>(&Bs[bn * BK + bkh])     = *reinterpret_cast<uint4*>(&hb[0]);
            *reinterpret_cast<uint4*>(&Bs[bn * BK + bkh + 8]) = *reinterpret_cast<uint4*>(&hb[8]);
        }
        __syncthreads();

        // ---- issue next tile's global loads (overlap with WMMA) ----
        if (k0 + BK < K) {
            const int kn = k0 + BK;
#pragma unroll
            for (int i = 0; i < 4; ++i)
                aR[i] = *reinterpret_cast<const float4*>(
                    &A[(size_t)(m0 + arow + 32 * i) * lda + kn + af4]);
            const float* wc = &W[(size_t)(kn + bkh) * N + n0 + bn];
#pragma unroll
            for (int j = 0; j < 16; ++j) bR[j] = wc[(size_t)j * N];
            __builtin_prefetch(&A[(size_t)(m0 + arow) * lda + kn]);
        }

        // ---- fragments per ISA layout ----
        // A 16x32 bf16: lane<16 row=lane K{0..7,16..23}; lane>=16 K{8..15,24..31}
        Frag16 af[2];
#pragma unroll
        for (int fm = 0; fm < 2; ++fm) {
            int mrow = wm * 32 + fm * 16 + l15;
            int lo   = lhalf * 8;
            af[fm].q[0] = *reinterpret_cast<const uint4*>(&As[mrow * BK + lo]);
            af[fm].q[1] = *reinterpret_cast<const uint4*>(&As[mrow * BK + 16 + lo]);
        }
        // B 32x16 bf16: lane<16 col=lane K{0..7,16..23}; lane>=16 K{8..15,24..31}
#pragma unroll
        for (int fn = 0; fn < 4; ++fn) {
            Frag16 bf_;
            int ncol = wn * 64 + fn * 16 + l15;
            int khi  = lhalf * 8;
            bf_.q[0] = *reinterpret_cast<const uint4*>(&Bs[ncol * BK + khi]);
            bf_.q[1] = *reinterpret_cast<const uint4*>(&Bs[ncol * BK + khi + 16]);
#pragma unroll
            for (int fm = 0; fm < 2; ++fm)
                acc[fm][fn] = __builtin_amdgcn_wmma_f32_16x16x32_bf16(
                    false, af[fm].v, false, bf_.v,
                    (short)0, acc[fm][fn], false, false);
        }
        __syncthreads();
    }

    // ---- fused epilogue (no guards: tile-aligned) ----
    // C/D layout: elem r -> M = r + 8*(lane>=16), N = lane&15
#pragma unroll
    for (int fm = 0; fm < 2; ++fm) {
#pragma unroll
        for (int fn = 0; fn < 4; ++fn) {
            const int n = n0 + wn * 64 + fn * 16 + l15;
            const float bn_ = bias ? bias[n] : 0.f;
#pragma unroll
            for (int r = 0; r < 8; ++r) {
                const int m = m0 + wm * 32 + fm * 16 + lhalf * 8 + r;
                float v = acc[fm][fn][r] + bn_;
                if (R)    v += rscale * R[(size_t)m * N + n];
                if (relu) v = fmaxf(v, 0.f);
                C[(size_t)m * N + n] = v;
            }
        }
    }
}

// ===========================================================================
// GENERIC PATH (guarded, A transposed): used only for the small per-level
// 1x1-conv projections where grd is stored channel-major: A[k*lda + m].
// Block tile 128x64x32, 8 waves; wave tile 32x32 = 2x2 WMMA frags.
// ===========================================================================
__global__ __launch_bounds__(256) void wmma_gemm_trans_kernel(
    const float* __restrict__ A, long long strideA, int lda,
    const float* __restrict__ W,
    const float* __restrict__ bias, long long strideC,
    float* __restrict__ C,
    int M, int N, int K)
{
    constexpr int BM = 128, BN = 64, BK = 32;
    __shared__ unsigned short As[BM * BK];
    __shared__ unsigned short Bs[BN * BK];

    const int tid = threadIdx.x;
    const float* Ab = A + (size_t)blockIdx.z * (size_t)strideA;
    float*       Cb = C + (size_t)blockIdx.z * (size_t)strideC;

    const int m0 = blockIdx.x * BM;
    const int n0 = blockIdx.y * BN;
    const int wave = tid >> 5, lane = tid & 31;
    const int wm = wave & 3, wn = wave >> 2;
    const int l15 = lane & 15, lhalf = (lane >> 4) & 1;

    v8f acc[2][2];
#pragma unroll
    for (int i = 0; i < 2; ++i)
#pragma unroll
        for (int j = 0; j < 2; ++j)
            acc[i][j] = (v8f){0.f, 0.f, 0.f, 0.f, 0.f, 0.f, 0.f, 0.f};

    for (int k0 = 0; k0 < K; k0 += BK) {
#pragma unroll
        for (int i = 0; i < (BM * BK) / 256; ++i) {
            int idx = i * 256 + tid;
            int m = idx & (BM - 1), k = idx >> 7;          // coalesced on m
            int gm = m0 + m, gk = k0 + k;
            float v = (gm < M && gk < K) ? Ab[(size_t)gk * lda + gm] : 0.f;
            As[m * BK + k] = f2bf(v);
        }
#pragma unroll
        for (int i = 0; i < (BN * BK) / 256; ++i) {
            int idx = i * 256 + tid;
            int n = idx & (BN - 1), k = idx >> 6;
            int gn = n0 + n, gk = k0 + k;
            float v = (gn < N && gk < K) ? W[(size_t)gk * N + gn] : 0.f;
            Bs[n * BK + k] = f2bf(v);
        }
        __syncthreads();

        Frag16 af[2], bf_[2];
#pragma unroll
        for (int fm = 0; fm < 2; ++fm) {
            int mrow = wm * 32 + fm * 16 + l15;
            int lo   = lhalf * 8;
            af[fm].q[0] = *reinterpret_cast<const uint4*>(&As[mrow * BK + lo]);
            af[fm].q[1] = *reinterpret_cast<const uint4*>(&As[mrow * BK + 16 + lo]);
        }
#pragma unroll
        for (int fn = 0; fn < 2; ++fn) {
            int ncol = wn * 32 + fn * 16 + l15;
            int khi  = lhalf * 8;
            bf_[fn].q[0] = *reinterpret_cast<const uint4*>(&Bs[ncol * BK + khi]);
            bf_[fn].q[1] = *reinterpret_cast<const uint4*>(&Bs[ncol * BK + khi + 16]);
        }
#pragma unroll
        for (int fm = 0; fm < 2; ++fm)
#pragma unroll
            for (int fn = 0; fn < 2; ++fn)
                acc[fm][fn] = __builtin_amdgcn_wmma_f32_16x16x32_bf16(
                    false, af[fm].v, false, bf_[fn].v,
                    (short)0, acc[fm][fn], false, false);
        __syncthreads();
    }

#pragma unroll
    for (int fm = 0; fm < 2; ++fm) {
#pragma unroll
        for (int fn = 0; fn < 2; ++fn) {
            int n = n0 + wn * 32 + fn * 16 + l15;
#pragma unroll
            for (int r = 0; r < 8; ++r) {
                int m = m0 + wm * 32 + fm * 16 + lhalf * 8 + r;
                if (m < M && n < N) {
                    float v = acc[fm][fn][r];
                    if (bias) v += bias[n];
                    Cb[(size_t)m * N + n] = v;
                }
            }
        }
    }
}

// ---------------------------------------------------------------------------
// GroupNorm over one (batch, group): 8 channels x HW pixels, in place.
// ---------------------------------------------------------------------------
__global__ __launch_bounds__(256) void groupnorm_kernel(
    float* __restrict__ value,
    const float* __restrict__ gamma, const float* __restrict__ beta,
    int HW, int start, float inv_cnt)
{
    __shared__ float s_sum[256];
    __shared__ float s_sq[256];
    const int b  = blockIdx.x >> 5;
    const int g  = blockIdx.x & 31;
    const int c0 = g * 8;
    float* base = value + ((size_t)b * NVTOT + start) * EMBED;
    const int cnt = HW * 8;

    float s = 0.f, sq = 0.f;
    for (int i = threadIdx.x; i < cnt; i += 256) {
        int pix = i >> 3, j = i & 7;
        float v = base[(size_t)pix * EMBED + c0 + j];
        s += v; sq += v * v;
    }
    s_sum[threadIdx.x] = s; s_sq[threadIdx.x] = sq;
    __syncthreads();
    for (int off = 128; off; off >>= 1) {
        if (threadIdx.x < off) {
            s_sum[threadIdx.x] += s_sum[threadIdx.x + off];
            s_sq[threadIdx.x]  += s_sq[threadIdx.x + off];
        }
        __syncthreads();
    }
    const float mean = s_sum[0] * inv_cnt;
    const float var  = s_sq[0] * inv_cnt - mean * mean;
    const float rstd = rsqrtf(var + 1e-5f);

    for (int i = threadIdx.x; i < cnt; i += 256) {
        int pix = i >> 3, j = i & 7, c = c0 + j;
        size_t o = (size_t)pix * EMBED + c;
        base[o] = (base[o] - mean) * rstd * gamma[c] + beta[c];
    }
}

// ---------------------------------------------------------------------------
// Softmax over the 16 (level,point) logits per (b, q, head), in place.
// ---------------------------------------------------------------------------
__global__ __launch_bounds__(256) void softmax16_kernel(float* __restrict__ aw, int total)
{
    int t = blockIdx.x * blockDim.x + threadIdx.x;
    if (t >= total) return;
    float* p = aw + (size_t)t * 16;
    float m = p[0];
#pragma unroll
    for (int i = 1; i < 16; ++i) m = fmaxf(m, p[i]);
    float e[16], s = 0.f;
#pragma unroll
    for (int i = 0; i < 16; ++i) { e[i] = __expf(p[i] - m); s += e[i]; }
    float inv = 1.f / s;
#pragma unroll
    for (int i = 0; i < 16; ++i) p[i] = e[i] * inv;
}

// ---------------------------------------------------------------------------
// Deformable sampling: one wave32 per (b, q, head); lane = head-dim channel.
// ---------------------------------------------------------------------------
__device__ __forceinline__ float fetchv(const float* __restrict__ vb,
                                        int xi, int yi, int D)
{
    if (xi < 0 || yi < 0 || xi >= D || yi >= D) return 0.f;
    return vb[(size_t)(yi * D + xi) * EMBED];
}

__global__ __launch_bounds__(256) void msdeform_sample_kernel(
    const float* __restrict__ v,      // (B, NVTOT, 256)
    const float* __restrict__ offs,   // (B*NQ, 256)
    const float* __restrict__ aw,     // (B*NQ, 128) post-softmax
    float* __restrict__ attn,         // (B*NQ, 256)
    int B)
{
    const int gwave = (blockIdx.x * blockDim.x + threadIdx.x) >> 5;
    const int lane  = threadIdx.x & 31;
    const int h = gwave & 7;
    const int t = gwave >> 3;
    const int q = t & (NQ - 1);
    const int b = t >> 14;
    if (b >= B) return;

    const int dims_[LEVELS]   = {64, 32, 16, 8};
    const int starts_[LEVELS] = {0, 4096, 5120, 5376};

    const float refx = (float)(q & 127) * (1.0f / 127.0f);
    const float refy = (float)(q >> 7) * (1.0f / 127.0f);

    const size_t row = (size_t)b * NQ + q;
    const float* offrow = offs + row * EMBED + h * 32;  // [l*8 + p*2 + xy]
    const float* awrow  = aw   + row * 128   + h * 16;  // [l*4 + p]

    float acc = 0.f;
#pragma unroll
    for (int l = 0; l < LEVELS; ++l) {
        const int   D  = dims_[l];
        const float fD = (float)D;
        const float* vbase =
            v + ((size_t)b * NVTOT + starts_[l]) * EMBED + h * HEAD_DIM + lane;
#pragma unroll
        for (int p = 0; p < POINTS; ++p) {
            float ox = offrow[l * 8 + p * 2 + 0];
            float oy = offrow[l * 8 + p * 2 + 1];
            float x = (refx + ox / fD) * fD - 0.5f;
            float y = (refy + oy / fD) * fD - 0.5f;
            float x0f = floorf(x), y0f = floorf(y);
            float dx = x - x0f, dy = y - y0f;
            int x0 = (int)x0f, y0 = (int)y0f;
            float g00 = fetchv(vbase, x0,     y0,     D);
            float g01 = fetchv(vbase, x0 + 1, y0,     D);
            float g10 = fetchv(vbase, x0,     y0 + 1, D);
            float g11 = fetchv(vbase, x0 + 1, y0 + 1, D);
            float bil = (g00 * (1.f - dx) + g01 * dx) * (1.f - dy)
                      + (g10 * (1.f - dx) + g11 * dx) * dy;
            acc += awrow[l * 4 + p] * bil;
        }
    }
    attn[row * EMBED + h * HEAD_DIM + lane] = acc;
}

// ---------------------------------------------------------------------------
// LayerNorm(ffn_pre) * g + b + fused -> out.  One wave32 per 256-wide row.
// ---------------------------------------------------------------------------
__global__ __launch_bounds__(256) void ln_residual_kernel(
    const float* __restrict__ ffn_pre,
    const float* __restrict__ fused,
    const float* __restrict__ g, const float* __restrict__ be,
    float* __restrict__ out, int rows)
{
    const int wave = threadIdx.x >> 5;
    const int lane = threadIdx.x & 31;
    const int row  = blockIdx.x * 8 + wave;
    if (row >= rows) return;
    const float* x = ffn_pre + (size_t)row * EMBED;

    float e[8], s = 0.f;
#pragma unroll
    for (int i = 0; i < 8; ++i) { e[i] = x[lane + i * 32]; s += e[i]; }
#pragma unroll
    for (int off = 16; off; off >>= 1) s += __shfl_xor(s, off, 32);
    const float mean = s * (1.0f / 256.0f);

    float vs = 0.f;
#pragma unroll
    for (int i = 0; i < 8; ++i) { float d = e[i] - mean; vs += d * d; }
#pragma unroll
    for (int off = 16; off; off >>= 1) vs += __shfl_xor(vs, off, 32);
    const float rstd = rsqrtf(vs * (1.0f / 256.0f) + 1e-5f);

#pragma unroll
    for (int i = 0; i < 8; ++i) {
        int c = lane + i * 32;
        out[(size_t)row * EMBED + c] =
            fused[(size_t)row * EMBED + c] + (e[i] - mean) * rstd * g[c] + be[c];
    }
}

// ---------------------------------------------------------------------------
// Host launcher
// ---------------------------------------------------------------------------
static void launch_gemm_aligned(const float* A, int lda, const float* W,
                                const float* bias, const float* R, float rscale,
                                float* C, int M, int N, int K, int relu,
                                hipStream_t s)
{
    dim3 grid(M / 128, N / 128, 1);
    wmma_gemm_aligned_kernel<<<grid, 256, 0, s>>>(A, lda, W, bias, R, rscale,
                                                  C, M, N, K, relu);
}

extern "C" void kernel_launch(void* const* d_in, const int* in_sizes, int n_in,
                              void* d_out, int out_size, void* d_ws, size_t ws_size,
                              hipStream_t stream)
{
    (void)n_in; (void)out_size; (void)ws_size;
    const float* Q    = (const float*)d_in[0];
    const float* grd[LEVELS] = {(const float*)d_in[1], (const float*)d_in[2],
                                (const float*)d_in[3], (const float*)d_in[4]};
    const float *proj_w[LEVELS], *proj_b[LEVELS], *gn_g[LEVELS], *gn_b[LEVELS];
    for (int l = 0; l < LEVELS; ++l) {
        proj_w[l] = (const float*)d_in[6 + l * 4 + 0];
        proj_b[l] = (const float*)d_in[6 + l * 4 + 1];
        gn_g[l]   = (const float*)d_in[6 + l * 4 + 2];
        gn_b[l]   = (const float*)d_in[6 + l * 4 + 3];
    }
    const float* value_w = (const float*)d_in[22];
    const float* value_b = (const float*)d_in[23];
    const float* off_w   = (const float*)d_in[24];
    const float* off_b   = (const float*)d_in[25];
    const float* attw_w  = (const float*)d_in[26];
    const float* attw_b  = (const float*)d_in[27];
    const float* out_w   = (const float*)d_in[28];
    const float* out_b   = (const float*)d_in[29];
    const float* lin1_w  = (const float*)d_in[30];
    const float* lin1_b  = (const float*)d_in[31];
    const float* lin2_w  = (const float*)d_in[32];
    const float* lin2_b  = (const float*)d_in[33];
    const float* ln_g    = (const float*)d_in[34];
    const float* ln_b    = (const float*)d_in[35];

    const int B  = in_sizes[0] / (NQ * EMBED);
    const int MQ = B * NQ;

    // workspace layout (floats)
    float* ws = (float*)d_ws;
    size_t off = 0;
    float* value  = ws + off; off += (size_t)B * NVTOT * EMBED;
    float* vproj  = ws + off; off += (size_t)B * NVTOT * EMBED;
    float* offs   = ws + off; off += (size_t)MQ * EMBED;
    float* aw     = ws + off; off += (size_t)MQ * 128;
    float* attnr  = ws + off; off += (size_t)MQ * EMBED;
    float* fused  = ws + off; off += (size_t)MQ * EMBED;
    float* hidden = ws + off; off += (size_t)MQ * HIDDEN;
    float* ffnp   = attnr;    // attnr dead after the out-proj GEMM

    const int HWs[LEVELS]    = {4096, 1024, 256, 64};
    const int Cs[LEVELS]     = {24, 40, 112, 1280};
    const int starts[LEVELS] = {0, 4096, 5120, 5376};

    // 1) per-level projection (grd is (C,HW): K-major -> transposed A) + bias
    for (int l = 0; l < LEVELS; ++l) {
        dim3 grid((HWs[l] + 127) / 128, EMBED / 64, B);
        wmma_gemm_trans_kernel<<<grid, 256, 0, stream>>>(
            grd[l], (long long)Cs[l] * HWs[l], HWs[l],
            proj_w[l], proj_b[l], (long long)NVTOT * EMBED,
            value + (size_t)starts[l] * EMBED,
            HWs[l], EMBED, Cs[l]);
    }
    // 2) GroupNorm(32) per level, in place
    for (int l = 0; l < LEVELS; ++l) {
        groupnorm_kernel<<<dim3(B * 32), 256, 0, stream>>>(
            value, gn_g[l], gn_b[l], HWs[l], starts[l], 1.0f / (HWs[l] * 8));
    }
    // 3) value projection  (M=B*5440=21760 — multiple of 128)
    launch_gemm_aligned(value, EMBED, value_w, value_b, nullptr, 0.f,
                        vproj, B * NVTOT, EMBED, EMBED, 0, stream);
    // 4) sampling offsets
    launch_gemm_aligned(Q, EMBED, off_w, off_b, nullptr, 0.f,
                        offs, MQ, EMBED, EMBED, 0, stream);
    // 5) attention weight logits (N=128)
    launch_gemm_aligned(Q, EMBED, attw_w, attw_b, nullptr, 0.f,
                        aw, MQ, 128, EMBED, 0, stream);
    // 6) softmax over (L*P)=16 per head
    {
        int total = MQ * HEADS;
        softmax16_kernel<<<dim3((total + 255) / 256), 256, 0, stream>>>(aw, total);
    }
    // 7) deformable bilinear sampling + weighted sum -> attn_raw
    {
        int waves = MQ * HEADS;                 // one wave32 each
        msdeform_sample_kernel<<<dim3(waves / 8), 256, 0, stream>>>(
            vproj, offs, aw, attnr, B);
    }
    // 8) output proj: fused = attnr @ out_w + out_b + 2*Q
    launch_gemm_aligned(attnr, EMBED, out_w, out_b, Q, 2.0f,
                        fused, MQ, EMBED, EMBED, 0, stream);
    // 9) FFN lin1 + ReLU
    launch_gemm_aligned(fused, EMBED, lin1_w, lin1_b, nullptr, 0.f,
                        hidden, MQ, HIDDEN, EMBED, 1, stream);
    // 10) FFN lin2
    launch_gemm_aligned(hidden, HIDDEN, lin2_w, lin2_b, nullptr, 0.f,
                        ffnp, MQ, EMBED, HIDDEN, 0, stream);
    // 11) LayerNorm + residual -> out
    ln_residual_kernel<<<dim3((MQ + 7) / 8), 256, 0, stream>>>(
        ffnp, fused, ln_g, ln_b, (float*)d_out, MQ);
}
